// EnhancedGAT_20461224198766
// MI455X (gfx1250) — compile-verified, matrix operand order checked by reading, not run on hardware
//
#include <hip/hip_runtime.h>
#include <hip/hip_bf16.h>

// ---------------- problem constants (from the reference) ----------------
static constexpr int N_NODES = 50000;
static constexpr int N_EDGES = 1600000;
static constexpr int F_IN    = 256;
static constexpr int HEADS   = 6;
static constexpr int CH      = 16;
static constexpr int HC      = HEADS * CH;   // 96
static constexpr int K_OUT   = 40;
static constexpr int K_PAD   = 48;           // 40 padded to 3 col-tiles
static constexpr int NE_TOT  = N_EDGES + N_NODES; // with self loops
static constexpr float NEG_SLOPE = 0.2f;
static constexpr float EPS = 1e-16f;
static constexpr int TPB = 256;

typedef _Float16 h8  __attribute__((ext_vector_type(8)));
typedef _Float16 h16 __attribute__((ext_vector_type(16)));
typedef float    f8  __attribute__((ext_vector_type(8)));

static inline int gsz(long long n) { return (int)((n + TPB - 1) / TPB); }

// ---------------- small utility kernels ----------------
__global__ void k_cvt_f16(const float* __restrict__ in, _Float16* __restrict__ out, int n) {
  int i = blockIdx.x * blockDim.x + threadIdx.x;
  if (i < n) out[i] = (_Float16)in[i];
}

__global__ void k_fill_f32(float* __restrict__ p, float v, int n) {
  int i = blockIdx.x * blockDim.x + threadIdx.x;
  if (i < n) p[i] = v;
}

__global__ void k_fill_u32(unsigned* __restrict__ p, unsigned v, int n) {
  int i = blockIdx.x * blockDim.x + threadIdx.x;
  if (i < n) p[i] = v;
}

// W is K x Nout (row-major). Build WT: NoutPad x K (row-major, f16, zero padded).
__global__ void k_build_wt(const float* __restrict__ W, _Float16* __restrict__ WT,
                           int K, int Nout, int NoutPad) {
  int i = blockIdx.x * blockDim.x + threadIdx.x;
  if (i >= NoutPad * K) return;
  int n = i / K, k = i - n * K;
  WT[i] = (n < Nout) ? (_Float16)W[(size_t)k * Nout + n] : (_Float16)0.0f;
}

// ---------------- WMMA GEMM: C[M x (NTN*16)] = A[M x K] * WT^T ----------------
// A : M x K f16 row-major (K multiple of 32).
// WT: (NTN*16) x K f16 row-major (B transposed; padded rows zeroed).
// Block = 8 waves. WT is staged once per block into LDS with
// global_load_async_to_lds_b128 (ASYNCcnt) and read back with ds_load_b128.
// Each wave computes one 16-row strip across all NTN col tiles. All NTN B
// fragments are loaded into distinct registers first, then the NTN WMMAs run
// back-to-back (avoids a ds_load -> s_wait_dscnt -> wmma serial chain).
template <int NTN>
__global__ void k_gemm_wmma(const _Float16* __restrict__ A,
                            const _Float16* __restrict__ BT,
                            float* __restrict__ Cd,
                            int M, int K, int Nout, int ldc) {
  extern __shared__ _Float16 sB[];   // NTN*16 rows x K halves

  // ---- cooperative async stage of W^T into LDS ----
  {
    const int totBytes = NTN * 16 * K * (int)sizeof(_Float16);
    const unsigned ldsBase = (unsigned)(size_t)(&sB[0]);   // low 32 bits = LDS byte addr
    for (int b = threadIdx.x * 16; b < totBytes; b += TPB * 16) {
      unsigned    lds = ldsBase + (unsigned)b;
      const char* gp  = (const char*)BT + b;
      asm volatile("global_load_async_to_lds_b128 %0, %1, off"
                   :: "v"(lds), "v"(gp) : "memory");
    }
    asm volatile("s_wait_asynccnt 0" ::: "memory");
  }
  __syncthreads();

  const int lane = threadIdx.x & 31;
  const int wave = threadIdx.x >> 5;
  const int tm   = blockIdx.x * (TPB / 32) + wave;   // row tile (wave-uniform)
  const int ntm  = M >> 4;
  if (tm >= ntm) return;                             // EXEC stays all-ones per wave

  const int r15 = lane & 15;
  const int kb  = (lane >> 4) << 3;                  // 0 or 8 (K sub-offset)
  const _Float16* arow = A + (size_t)(tm * 16 + r15) * K;

  f8 acc[NTN];
#pragma unroll
  for (int t = 0; t < NTN; ++t) acc[t] = (f8){};

  for (int kc = 0; kc < K; kc += 32) {
    h8 alo = *(const h8*)(arow + kc + kb);
    h8 ahi = *(const h8*)(arow + kc + 16 + kb);
    h16 a  = __builtin_shufflevector(alo, ahi, 0,1,2,3,4,5,6,7,8,9,10,11,12,13,14,15);

    // batch-load all B fragments into distinct registers (one ds clause) ...
    h16 bf[NTN];
#pragma unroll
    for (int t = 0; t < NTN; ++t) {
      const _Float16* brow = sB + (size_t)(t * 16 + r15) * K + kc + kb;
      h8 blo = *(const h8*)(brow);
      h8 bhi = *(const h8*)(brow + 16);
      bf[t]  = __builtin_shufflevector(blo, bhi, 0,1,2,3,4,5,6,7,8,9,10,11,12,13,14,15);
    }
    // ... then issue the WMMAs back-to-back
#pragma unroll
    for (int t = 0; t < NTN; ++t) {
      acc[t] = __builtin_amdgcn_wmma_f32_16x16x32_f16(false, a, false, bf[t], (short)0,
                                                      acc[t], false, false);
    }
  }

  const int rbase = tm * 16 + ((lane >> 4) << 3);
#pragma unroll
  for (int t = 0; t < NTN; ++t) {
    const int col = t * 16 + r15;
    if (col < Nout) {
#pragma unroll
      for (int r = 0; r < 8; ++r)
        Cd[(size_t)(rbase + r) * ldc + col] = acc[t][r];
    }
  }
}

// ---------------- attention logits: ls/ld[n,h] = <h[n,h,:], a_{s,d}[h,:]> ----------------
__global__ void k_logits(const float* __restrict__ hf, const float* __restrict__ as,
                         const float* __restrict__ ad, float* __restrict__ ls,
                         float* __restrict__ ldv, int Nn, int Hh, int Cc) {
  int i = blockIdx.x * blockDim.x + threadIdx.x;
  if (i >= Nn * Hh) return;
  int n = i / Hh, hh = i - n * Hh;
  const float* row = hf + (size_t)n * Hh * Cc + hh * Cc;
  float s = 0.f, d = 0.f;
  for (int c = 0; c < Cc; ++c) {
    s = fmaf(row[c], as[hh * Cc + c], s);
    d = fmaf(row[c], ad[hh * Cc + c], d);
  }
  ls[i] = s;
  ldv[i] = d;
}

// ---------------- ordered-uint encoding for float atomic max ----------------
__device__ __forceinline__ unsigned ordf(float f) {
  unsigned u = __float_as_uint(f);
  return (u >> 31) ? ~u : (u | 0x80000000u);
}
__device__ __forceinline__ float unordf(unsigned u) {
  unsigned b = (u & 0x80000000u) ? (u ^ 0x80000000u) : ~u;
  return __uint_as_float(b);
}

__device__ __forceinline__ void edge_sd(int e, const int* __restrict__ src,
                                        const int* __restrict__ dst, int& s, int& d) {
  if (e < N_EDGES) { s = src[e]; d = dst[e]; }
  else             { s = e - N_EDGES; d = s; }   // self loop
}

// pass 1: segment max of leaky_relu(ls[src]+ld[dst])
__global__ void k_edge_max(const int* __restrict__ src, const int* __restrict__ dst,
                           const float* __restrict__ ls, const float* __restrict__ ldv,
                           unsigned* __restrict__ mmax, int Hh) {
  int i = blockIdx.x * blockDim.x + threadIdx.x;
  if (i >= NE_TOT * Hh) return;
  int e = i / Hh, hh = i - e * Hh;
  int s, d; edge_sd(e, src, dst, s, d);
  float v = ls[s * Hh + hh] + ldv[d * Hh + hh];
  v = (v > 0.f) ? v : v * NEG_SLOPE;
  atomicMax(&mmax[d * Hh + hh], ordf(v));
}

// pass 2: ex = exp(e - m[dst]); denom[dst] += ex
__global__ void k_edge_exp(const int* __restrict__ src, const int* __restrict__ dst,
                           const float* __restrict__ ls, const float* __restrict__ ldv,
                           const unsigned* __restrict__ mmax, float* __restrict__ ex,
                           float* __restrict__ denom, int Hh) {
  int i = blockIdx.x * blockDim.x + threadIdx.x;
  if (i >= NE_TOT * Hh) return;
  int e = i / Hh, hh = i - e * Hh;
  int s, d; edge_sd(e, src, dst, s, d);
  float v = ls[s * Hh + hh] + ldv[d * Hh + hh];
  v = (v > 0.f) ? v : v * NEG_SLOPE;
  float m = unordf(mmax[d * Hh + hh]);
  float x = __expf(v - m);
  ex[i] = x;
  atomicAdd(&denom[d * Hh + hh], x);
}

// pass 3: agg[dst,h,:] += (ex / (denom+eps)) * h[src,h,:]
__global__ void k_edge_agg(const int* __restrict__ src, const int* __restrict__ dst,
                           const float* __restrict__ ex, const float* __restrict__ denom,
                           const float* __restrict__ hf, float* __restrict__ agg,
                           int Hh, int Cc) {
  int i = blockIdx.x * blockDim.x + threadIdx.x;
  if (i >= NE_TOT * Hh) return;
  int e = i / Hh, hh = i - e * Hh;
  int s, d; edge_sd(e, src, dst, s, d);
  float coeff = ex[i] / (denom[d * Hh + hh] + EPS);
  const float* hr = hf  + (size_t)s * Hh * Cc + hh * Cc;
  float*       ar = agg + (size_t)d * Hh * Cc + hh * Cc;
  for (int c = 0; c < Cc; ++c) atomicAdd(&ar[c], coeff * hr[c]);
}

// bias + optional relu, in place
__global__ void k_bias_act(float* __restrict__ agg, const float* __restrict__ b,
                           int Nn, int Fw, int do_relu) {
  int i = blockIdx.x * blockDim.x + threadIdx.x;
  if (i >= Nn * Fw) return;
  int j = i % Fw;
  float v = agg[i] + b[j];
  if (do_relu) v = fmaxf(v, 0.f);
  agg[i] = v;
}

// final: out[n,:] = log_softmax(agg[n,:] + b)
__global__ void k_logsoftmax(const float* __restrict__ agg, const float* __restrict__ b,
                             float* __restrict__ out, int Nn) {
  int n = blockIdx.x * blockDim.x + threadIdx.x;
  if (n >= Nn) return;
  float v[K_OUT];
  float mx = -3.402823466e38f;
  for (int j = 0; j < K_OUT; ++j) {
    v[j] = agg[(size_t)n * K_OUT + j] + b[j];
    mx = fmaxf(mx, v[j]);
  }
  float s = 0.f;
  for (int j = 0; j < K_OUT; ++j) s += __expf(v[j] - mx);
  float lse = __logf(s);
  for (int j = 0; j < K_OUT; ++j) out[(size_t)n * K_OUT + j] = v[j] - mx - lse;
}

// ---------------- host orchestration ----------------
static inline size_t al256(size_t x) { return (x + 255) & ~(size_t)255; }

extern "C" void kernel_launch(void* const* d_in, const int* in_sizes, int n_in,
                              void* d_out, int out_size, void* d_ws, size_t ws_size,
                              hipStream_t stream) {
  (void)in_sizes; (void)n_in; (void)out_size; (void)ws_size;
  const float* x    = (const float*)d_in[0];
  const int*   ei   = (const int*)d_in[1];
  const int*   esrc = ei;
  const int*   edst = ei + N_EDGES;
  const float* W[4]  = {(const float*)d_in[2],  (const float*)d_in[6],
                        (const float*)d_in[10], (const float*)d_in[14]};
  const float* AS[4] = {(const float*)d_in[3],  (const float*)d_in[7],
                        (const float*)d_in[11], (const float*)d_in[15]};
  const float* AD[4] = {(const float*)d_in[4],  (const float*)d_in[8],
                        (const float*)d_in[12], (const float*)d_in[16]};
  const float* B[4]  = {(const float*)d_in[5],  (const float*)d_in[9],
                        (const float*)d_in[13], (const float*)d_in[17]};
  float* out = (float*)d_out;

  // ---- carve workspace ----
  char* w = (char*)d_ws;
  size_t off = 0;
  _Float16* hf16 = (_Float16*)(w + off); off += al256((size_t)N_NODES * F_IN * 2);   // 25.6 MB
  _Float16* wt16 = (_Float16*)(w + off); off += al256((size_t)HC * F_IN * 2);        // 48 KB (max)
  float*    hW   = (float*)(w + off);    off += al256((size_t)N_NODES * HC * 4);     // 19.2 MB
  float*    agg  = (float*)(w + off);    off += al256((size_t)N_NODES * HC * 4);     // 19.2 MB
  float*    ls   = (float*)(w + off);    off += al256((size_t)N_NODES * HEADS * 4);
  float*    ldv  = (float*)(w + off);    off += al256((size_t)N_NODES * HEADS * 4);
  unsigned* mmax = (unsigned*)(w + off); off += al256((size_t)N_NODES * HEADS * 4);
  float*    den  = (float*)(w + off);    off += al256((size_t)N_NODES * HEADS * 4);
  float*    ex   = (float*)(w + off);    off += al256((size_t)NE_TOT * HEADS * 4);   // 39.6 MB

  const float* feat = x;   // current fp32 node features
  int fin = F_IN;

  const int ntm       = N_NODES / 16;              // 3125 row tiles
  const int gemmBlks  = (ntm + 7) / 8;             // 8 waves per block

  // ---------------- layers 1..3 (H=6, C=16, concat, relu) ----------------
  for (int l = 0; l < 3; ++l) {
    // fp32 -> fp16 features
    k_cvt_f16<<<gsz((long long)N_NODES * fin), TPB, 0, stream>>>(feat, hf16,
                                                                 N_NODES * fin);
    // W^T in f16 (96 x fin)
    k_build_wt<<<gsz((long long)HC * fin), TPB, 0, stream>>>(W[l], wt16, fin, HC, HC);
    // hW = feat @ W   via WMMA, W^T async-staged in LDS
    {
      size_t shmem = (size_t)HC * fin * sizeof(_Float16);
      k_gemm_wmma<6><<<gemmBlks, TPB, shmem, stream>>>(hf16, wt16, hW, N_NODES, fin,
                                                       HC, HC);
    }
    // logits
    k_logits<<<gsz((long long)N_NODES * HEADS), TPB, 0, stream>>>(
        hW, AS[l], AD[l], ls, ldv, N_NODES, HEADS, CH);
    // init reductions
    k_fill_u32<<<gsz((long long)N_NODES * HEADS), TPB, 0, stream>>>(mmax, 0u,
                                                                    N_NODES * HEADS);
    k_fill_f32<<<gsz((long long)N_NODES * HEADS), TPB, 0, stream>>>(den, 0.f,
                                                                    N_NODES * HEADS);
    k_fill_f32<<<gsz((long long)N_NODES * HC), TPB, 0, stream>>>(agg, 0.f,
                                                                 N_NODES * HC);
    // edge passes
    k_edge_max<<<gsz((long long)NE_TOT * HEADS), TPB, 0, stream>>>(
        esrc, edst, ls, ldv, mmax, HEADS);
    k_edge_exp<<<gsz((long long)NE_TOT * HEADS), TPB, 0, stream>>>(
        esrc, edst, ls, ldv, mmax, ex, den, HEADS);
    k_edge_agg<<<gsz((long long)NE_TOT * HEADS), TPB, 0, stream>>>(
        esrc, edst, ex, den, hW, agg, HEADS, CH);
    // bias + relu (in place); agg becomes next layer's input
    k_bias_act<<<gsz((long long)N_NODES * HC), TPB, 0, stream>>>(agg, B[l], N_NODES,
                                                                 HC, 1);
    feat = agg;
    fin = HC;
  }

  // ---------------- layer 4 (H=1, C=40, mean==identity, log_softmax) ----------------
  k_cvt_f16<<<gsz((long long)N_NODES * HC), TPB, 0, stream>>>(feat, hf16,
                                                              N_NODES * HC);
  k_build_wt<<<gsz((long long)K_PAD * HC), TPB, 0, stream>>>(W[3], wt16, HC, K_OUT,
                                                             K_PAD);
  {
    size_t shmem = (size_t)K_PAD * HC * sizeof(_Float16);
    k_gemm_wmma<3><<<gemmBlks, TPB, shmem, stream>>>(hf16, wt16, hW, N_NODES, HC,
                                                     K_OUT, K_OUT); // hW: N x 40
  }
  k_logits<<<gsz((long long)N_NODES), TPB, 0, stream>>>(hW, AS[3], AD[3], ls, ldv,
                                                        N_NODES, 1, K_OUT);
  k_fill_u32<<<gsz((long long)N_NODES), TPB, 0, stream>>>(mmax, 0u, N_NODES);
  k_fill_f32<<<gsz((long long)N_NODES), TPB, 0, stream>>>(den, 0.f, N_NODES);
  k_fill_f32<<<gsz((long long)N_NODES * K_OUT), TPB, 0, stream>>>(agg, 0.f,
                                                                  N_NODES * K_OUT);
  k_edge_max<<<gsz((long long)NE_TOT), TPB, 0, stream>>>(esrc, edst, ls, ldv, mmax, 1);
  k_edge_exp<<<gsz((long long)NE_TOT), TPB, 0, stream>>>(esrc, edst, ls, ldv, mmax, ex,
                                                         den, 1);
  k_edge_agg<<<gsz((long long)NE_TOT), TPB, 0, stream>>>(esrc, edst, ex, den, hW, agg,
                                                         1, K_OUT);
  k_logsoftmax<<<gsz((long long)N_NODES), TPB, 0, stream>>>(agg, B[3], out, N_NODES);
}
